// QLinear_40973988004347
// MI455X (gfx1250) — 8-bit matrix operand order repaired and checked by reading, compiled, not run on hardware
//
#include <hip/hip_runtime.h>
#include <cstdint>

typedef __attribute__((ext_vector_type(8))) int   v8i;
typedef __attribute__((ext_vector_type(4))) int   v4i;
typedef __attribute__((ext_vector_type(2))) int   v2i;
typedef __attribute__((ext_vector_type(4))) float v4f;

#define QMAX_F 127.0f

#define AS1 __attribute__((address_space(1)))
#define AS3 __attribute__((address_space(3)))

#if defined(__has_builtin)
#if __has_builtin(__builtin_amdgcn_global_load_async_to_lds_b128)
#define HAVE_ASYNC_LDS 1
#endif
#endif

#ifndef HAVE_ASYNC_LDS
#define HAVE_ASYNC_LDS 0
#endif

__device__ __forceinline__ void wait_async0() {
#if defined(__has_builtin) && __has_builtin(__builtin_amdgcn_s_wait_asynccnt)
    __builtin_amdgcn_s_wait_asynccnt(0);
#else
    asm volatile("s_wait_asynccnt 0x0" ::: "memory");
#endif
}

#if HAVE_ASYNC_LDS
__device__ __forceinline__ void async_copy16(const int8_t* g, int8_t* l) {
    __builtin_amdgcn_global_load_async_to_lds_b128(
        (AS1 v4i*)(uintptr_t)g,
        (AS3 v4i*)(uint32_t)(uintptr_t)l,
         0,  0);
}
#endif

__device__ __forceinline__ int clamp_q(float v) {
    int q = __float2int_rn(v);
    q = q > 127 ? 127 : q;
    q = q < -127 ? -127 : q;
    return q;
}

__global__ __launch_bounds__(256) void quant_rows(const float* __restrict__ x,
                                                  int8_t* __restrict__ qx,
                                                  float* __restrict__ iscale,
                                                  int K) {
    const int row  = blockIdx.x * 8 + (threadIdx.x >> 5);
    const int lane = threadIdx.x & 31;
    const float* xr = x + (size_t)row * K;
    const int nf4 = K >> 2;

    float amax = 0.0f;
    for (int f = lane; f < nf4; f += 32) {
        v4f v = *(const v4f*)(xr + 4 * f);
        float m0 = fmaxf(fabsf(v.x), fabsf(v.y));
        float m1 = fmaxf(fabsf(v.z), fabsf(v.w));
        amax = fmaxf(amax, fmaxf(m0, m1));
    }
#pragma unroll
    for (int off = 16; off > 0; off >>= 1)
        amax = fmaxf(amax, __shfl_xor(amax, off, 32));

    const float sc  = fmaxf(amax, 1e-8f) / QMAX_F;
    const float inv = 1.0f / sc;
    if (lane == 0) iscale[row] = sc;

    int* qo = (int*)(qx + (size_t)row * K);
    for (int f = lane; f < nf4; f += 32) {
        v4f v = *(const v4f*)(xr + 4 * f);
        int a = clamp_q(v.x * inv);
        int b = clamp_q(v.y * inv);
        int c = clamp_q(v.z * inv);
        int d = clamp_q(v.w * inv);
        qo[f] = (a & 255) | ((b & 255) << 8) | ((c & 255) << 16) | ((d & 255) << 24);
    }
}

__global__ __launch_bounds__(256) void pack_w(const int* __restrict__ qw,
                                              int8_t* __restrict__ wq) {
    const size_t i = (size_t)blockIdx.x * 256 + threadIdx.x;
    v4i v = *(const v4i*)(qw + 4 * i);
    ((int*)wq)[i] = (v.x & 255) | ((v.y & 255) << 8) | ((v.z & 255) << 16) | ((v.w & 255) << 24);
}

#define MB 128
#define NB 128
#define KS 64
#define LSTR 80

__global__ __launch_bounds__(256) void gemm_i8_wmma(const int8_t* __restrict__ qx,
                                                    const int8_t* __restrict__ wq,
                                                    const float* __restrict__ iscale,
                                                    const float* __restrict__ wscale,
                                                    const float* __restrict__ bias,
                                                    float* __restrict__ out,
                                                    int M, int N, int K) {
    __shared__ int8_t lA[2][MB * LSTR];
    __shared__ int8_t lB[2][NB * LSTR];

    const int tid  = threadIdx.x;
    const int lane = tid & 31;
    const int wave = tid >> 5;
    const int wm   = wave >> 1;
    const int wn   = wave & 1;

    const int m0 = blockIdx.y * MB;
    const int n0 = blockIdx.x * NB;

    auto stage = [&](int buf, int kbase) {
#pragma unroll
        for (int s = 0; s < 2; ++s) {
            const int seg = tid + s * 256;
            const int row = seg >> 2;
            const int col = (seg & 3) * 16;
            const int8_t* ga = qx + (size_t)(m0 + row) * K + kbase + col;
            const int8_t* gb = wq + (size_t)(n0 + row) * K + kbase + col;
            int8_t* la = &lA[buf][row * LSTR + col];
            int8_t* lb = &lB[buf][row * LSTR + col];
#if HAVE_ASYNC_LDS
            async_copy16(ga, la);
            async_copy16(gb, lb);
#else
            *(v4i*)la = *(const v4i*)ga;
            *(v4i*)lb = *(const v4i*)gb;
#endif
        }
    };

    auto loadA = [&](int buf, int mt) -> v8i {
        const int m    = wm * 32 + mt * 16 + (lane & 15);
        const int koff = (lane >> 4) * 8;
        const int8_t* p = &lA[buf][m * LSTR + koff];
        v2i a0 = *(const v2i*)(p + 0);
        v2i a1 = *(const v2i*)(p + 16);
        v2i a2 = *(const v2i*)(p + 32);
        v2i a3 = *(const v2i*)(p + 48);
        v8i r = {a0.x, a0.y, a1.x, a1.y, a2.x, a2.y, a3.x, a3.y};
        return r;
    };

    auto loadB = [&](int buf, int nt) -> v8i {
        const int n    = wn * 64 + nt * 16 + (lane & 15);
        const int koff = (lane >> 4) * 8;
        const int8_t* p = &lB[buf][n * LSTR + koff];
        v2i b0 = *(const v2i*)(p + 0);
        v2i b1 = *(const v2i*)(p + 16);
        v2i b2 = *(const v2i*)(p + 32);
        v2i b3 = *(const v2i*)(p + 48);
        v8i r = {b0.x, b0.y, b1.x, b1.y, b2.x, b2.y, b3.x, b3.y};
        return r;
    };

    v8i acc[2][4];
#pragma unroll
    for (int i = 0; i < 2; ++i)
#pragma unroll
        for (int j = 0; j < 4; ++j) acc[i][j] = (v8i)(0);

    const int ksteps = K / KS;

    stage(0, 0);
#if HAVE_ASYNC_LDS
    wait_async0();
#endif
    __syncthreads();

    for (int kt = 0; kt < ksteps; ++kt) {
        const int cur = kt & 1;
        const bool pf = (kt + 1 < ksteps);
        if (pf) stage(cur ^ 1, (kt + 1) * KS);

        v8i a[2], b[4];
#pragma unroll
        for (int i = 0; i < 2; ++i) a[i] = loadA(cur, i);
#pragma unroll
        for (int j = 0; j < 4; ++j) b[j] = loadB(cur, j);

#pragma unroll
        for (int i = 0; i < 2; ++i)
#pragma unroll
            for (int j = 0; j < 4; ++j)
                acc[i][j] = __builtin_amdgcn_wmma_i32_16x16x64_iu8(
                     true, a[i],  true, b[j], acc[i][j],
                     false,  false);

#if HAVE_ASYNC_LDS
        if (pf) wait_async0();
#endif
        __syncthreads();
    }

    const int coln  = lane & 15;
    const int rbase = (lane >> 4) * 8;
#pragma unroll
    for (int i = 0; i < 2; ++i) {
#pragma unroll
        for (int j = 0; j < 4; ++j) {
            const int n  = n0 + wn * 64 + j * 16 + coln;
            const float wsc = wscale[n];
            const float bb  = bias[n];
            const int mbase = m0 + wm * 32 + i * 16 + rbase;
#pragma unroll
            for (int r = 0; r < 8; ++r) {
                const int m = mbase + r;
                out[(size_t)m * N + n] = (float)acc[i][j][r] * iscale[m] * wsc + bb;
            }
        }
    }
}

extern "C" void kernel_launch(void* const* d_in, const int* in_sizes, int n_in,
                              void* d_out, int out_size, void* d_ws, size_t ws_size,
                              hipStream_t stream) {
    const float* inp     = (const float*)d_in[0];
    const int*   qweight = (const int*)d_in[1];
    const float* wscale  = (const float*)d_in[2];
    const float* bias    = (const float*)d_in[3];
    float*       out     = (float*)d_out;

    const int       N  = in_sizes[2];
    const long long NK = (long long)in_sizes[1];
    const int       K  = (int)(NK / N);
    const int       M  = (int)((long long)in_sizes[0] / K);

    int8_t* qx     = (int8_t*)d_ws;
    int8_t* wq     = qx + (size_t)M * K;
    float*  iscale = (float*)(wq + (size_t)N * K);

    quant_rows<<<M / 8, 256, 0, stream>>>(inp, qx, iscale, K);
    pack_w<<<(int)(NK / 1024), 256, 0, stream>>>(qweight, wq);

    dim3 grid(N / NB, M / MB);
    gemm_i8_wmma<<<grid, 256, 0, stream>>>(qx, wq, iscale, wscale, bias, out, M, N, K);
}
